// LinearAttention_9242769620996
// MI455X (gfx1250) — matrix operand order repaired and checked by reading, not compiled, not run on hardware
//
#include <hip/hip_runtime.h>
#include <hip/hip_bf16.h>

// ---------------------------------------------------------------------------
// Problem constants (from reference): B=8, N=4096, C=512, H=8, D=64
// ---------------------------------------------------------------------------
#define B_   8
#define N_   4096
#define C_   512
#define H_   8
#define D_   64
#define M_   (B_ * N_)          // 32768 rows
#define SCALE_ 0.015625f        // N^-0.5 = 1/64

typedef __attribute__((ext_vector_type(8)))  float  v8f;
typedef __attribute__((ext_vector_type(16))) __bf16 v16bf;

struct Bits256 { uint4 lo; uint4 hi; };

// float <-> bf16 bit helpers
__device__ __forceinline__ unsigned short f2bf(float f) {
    return __builtin_bit_cast(unsigned short, static_cast<__bf16>(f));
}
__device__ __forceinline__ float bf2f(unsigned short u) {
    return static_cast<float>(__builtin_bit_cast(__bf16, u));
}
// two floats -> packed bf16 pair (lowers to v_cvt_pk_bf16_f32)
__device__ __forceinline__ unsigned int pack_bf2(float lo, float hi) {
    return (unsigned int)f2bf(lo) | ((unsigned int)f2bf(hi) << 16);
}

// A-matrix fragment (16x32 bf16, row-major in LDS, row stride 32 elems).
// lane L: row = L&15, K(i) = (L>>4)*8 + (i<8 ? i : i+8)
__device__ __forceinline__ v16bf load_frag_a(const unsigned short* p, int lane) {
    int row = lane & 15;
    int kb  = (lane >> 4) << 3;
    const unsigned short* q = p + row * 32 + kb;
    Bits256 s;
    s.lo = *reinterpret_cast<const uint4*>(q);
    s.hi = *reinterpret_cast<const uint4*>(q + 16);
    return __builtin_bit_cast(v16bf, s);
}

// B-matrix fragment (32x16 bf16 = K x N), stored in LDS transposed as
// [n][K] row-major (row stride 32). lane L: col = L&15, K(i) = (L>>4)*8 + (i<8 ? i : i+8)
__device__ __forceinline__ v16bf load_frag_b(const unsigned short* p, int lane) {
    int n  = lane & 15;
    int kb = (lane >> 4) << 3;
    const unsigned short* q = p + n * 32 + kb;
    Bits256 s;
    s.lo = *reinterpret_cast<const uint4*>(q);
    s.hi = *reinterpret_cast<const uint4*>(q + 16);
    return __builtin_bit_cast(v16bf, s);
}

// ---------------------------------------------------------------------------
// Kernel 1/2: fused projection GEMM  Y = act(X @ W + bias)
// Block tile 128x128, 8 waves; double-buffered LDS, global->reg->LDS software
// pipeline so next-panel HBM latency overlaps the 8-WMMA chain.
//   mode 0: Q  (Nout=512):   y = elu(x)+1            -> q_bf (b,n,h,d) bf16
//   mode 1: KV (Nout=1024):  cols<512: k=elu(x)+1    -> kT_bf (bh,d,n)*1/64
//                            cols>=512: v            -> v_bf  (b,n,c)*1/64
// ---------------------------------------------------------------------------
__global__ __launch_bounds__(256) void proj_kernel(
    const float* __restrict__ X, const float* __restrict__ W,
    const float* __restrict__ bias, int Nout, int mode,
    unsigned short* __restrict__ q_bf,
    unsigned short* __restrict__ kT_bf,
    unsigned short* __restrict__ v_bf)
{
    __shared__ __align__(16) unsigned int smA[2][128 * 16];  // 128 rows x 32 bf16
    __shared__ __align__(16) unsigned int smB[2][128 * 16];  // 128 n    x 32 k

    const int tid  = threadIdx.x;
    const int lane = tid & 31;
    const int wave = tid >> 5;
    const int m0   = blockIdx.x * 128;
    const int n0   = blockIdx.y * 128;

    float2 ra[8];
    float  rb0[8], rb1[8];

    auto gload = [&](int k0) {
        #pragma unroll
        for (int j = 0; j < 8; ++j) {
            int p = tid + (j << 8);
            int r = p >> 4, c2 = p & 15;
            ra[j] = *reinterpret_cast<const float2*>(
                &X[(size_t)(m0 + r) * C_ + k0 + (c2 << 1)]);
        }
        #pragma unroll
        for (int j = 0; j < 8; ++j) {
            int p = tid + (j << 8);
            int c = p & 127, k2 = p >> 7;
            rb0[j] = W[(size_t)(k0 + (k2 << 1)    ) * Nout + n0 + c];
            rb1[j] = W[(size_t)(k0 + (k2 << 1) + 1) * Nout + n0 + c];
        }
    };
    auto lstore = [&](int buf) {
        #pragma unroll
        for (int j = 0; j < 8; ++j) {
            int p = tid + (j << 8);
            int r = p >> 4, c2 = p & 15;
            smA[buf][r * 16 + c2] = pack_bf2(ra[j].x, ra[j].y);
        }
        #pragma unroll
        for (int j = 0; j < 8; ++j) {
            int p = tid + (j << 8);
            int c = p & 127, k2 = p >> 7;
            smB[buf][c * 16 + k2] = pack_bf2(rb0[j], rb1[j]);
        }
    };

    v8f acc[8] = {{}, {}, {}, {}, {}, {}, {}, {}};

    gload(0);
    lstore(0);
    __syncthreads();

    int cur = 0;
    for (int k0 = 0; k0 < C_; k0 += 32) {
        const bool more = (k0 + 32) < C_;
        if (more) {
            gload(k0 + 32);                       // LOADcnt traffic, overlaps WMMA
            __builtin_prefetch(
                &X[(size_t)(m0 + (tid >> 1)) * C_ + k0 + 64], 0, 3);
        }

        v16bf a = load_frag_a(
            reinterpret_cast<const unsigned short*>(smA[cur]) + wave * 16 * 32,
            lane);
        #pragma unroll
        for (int t = 0; t < 8; ++t) {
            v16bf bb = load_frag_b(
                reinterpret_cast<const unsigned short*>(smB[cur]) + t * 16 * 32,
                lane);
            acc[t] = __builtin_amdgcn_wmma_f32_16x16x32_bf16(
                         false, a, false, bb, (short)0, acc[t], false, false);
        }

        if (more) lstore(cur ^ 1);                // waits LOADcnt, then ds_store
        __syncthreads();
        cur ^= 1;
    }

    // epilogue
    const int cl  = lane & 15;
    const int hi8 = (lane >> 4) << 3;
    #pragma unroll
    for (int t = 0; t < 8; ++t) {
        const int   col = n0 + t * 16 + cl;
        const float bv  = bias[col];
        #pragma unroll
        for (int r = 0; r < 8; ++r) {
            int   m = m0 + wave * 16 + r + hi8;   // global row in [0, B*N)
            float x = acc[t][r] + bv;
            if (mode == 0) {
                float y = x > 0.f ? x + 1.f : __expf(x);    // elu(x)+1
                q_bf[(size_t)m * C_ + col] = f2bf(y);
            } else {
                if (col < C_) {
                    float y  = x > 0.f ? x + 1.f : __expf(x);
                    int b  = m >> 12;
                    int n  = m & (N_ - 1);
                    int h  = col >> 6;
                    int dd = col & 63;
                    kT_bf[(((size_t)((b * H_ + h) * D_ + dd)) << 12) + n] =
                        f2bf(y * SCALE_);
                } else {
                    v_bf[(size_t)m * C_ + (col - C_)] = f2bf(x * SCALE_);
                }
            }
        }
    }
}

// ---------------------------------------------------------------------------
// Kernel 3: k_mean[bh*64+dd] = (1/N) * sum_n k = (1/64) * sum_n k_scaled
// ---------------------------------------------------------------------------
__global__ __launch_bounds__(256) void ksum_kernel(
    const unsigned short* __restrict__ kT, float* __restrict__ k_mean)
{
    __shared__ float red[256];
    const int row = blockIdx.x;                   // 0..4095 = bh*64+dd
    const unsigned short* p = kT + ((size_t)row << 12);
    float s = 0.f;
    for (int i = threadIdx.x; i < N_; i += 256) s += bf2f(p[i]);
    red[threadIdx.x] = s;
    __syncthreads();
    for (int off = 128; off > 0; off >>= 1) {
        if (threadIdx.x < off) red[threadIdx.x] += red[threadIdx.x + off];
        __syncthreads();
    }
    if (threadIdx.x == 0) k_mean[row] = red[0] * (1.0f / 64.0f);
}

// ---------------------------------------------------------------------------
// Kernel 4: z[bh*N+n] = 1 / (dot(q[b,n,h,:], k_mean[bh,:]) + 1e-6)
// ---------------------------------------------------------------------------
__global__ __launch_bounds__(256) void z_kernel(
    const unsigned short* __restrict__ q_bf,
    const float* __restrict__ k_mean, float* __restrict__ z)
{
    int t  = blockIdx.x * 256 + threadIdx.x;      // 0..B*H*N-1
    int n  = t & (N_ - 1);
    int bh = t >> 12;
    int b  = bh >> 3, h = bh & 7;
    const unsigned short* qp = q_bf + ((size_t)(b * N_ + n)) * C_ + h * D_;
    const float* km = k_mean + bh * D_;
    float s = 0.f;
    #pragma unroll 8
    for (int d = 0; d < D_; ++d) s += bf2f(qp[d]) * km[d];
    z[t] = 1.0f / (s + 1e-6f);
}

// ---------------------------------------------------------------------------
// Kernel 5: kv_state[bh] = (k*scale)^T @ (v*scale)   (64x64, K = 4096)
// 64 blocks (one per bh), 16 waves; block-shared panels, double-buffered,
// software-pipelined: kT and v each read exactly once from HBM.
// ---------------------------------------------------------------------------
__global__ __launch_bounds__(512) void kvstate_kernel(
    const unsigned short* __restrict__ kT,
    const unsigned short* __restrict__ v_bf,
    float* __restrict__ kv_state)
{
    __shared__ __align__(16) unsigned int smA[2][64 * 16]; // 64 d-rows x 32 k
    __shared__ __align__(16) unsigned int smB[2][64 * 16]; // 64 e-rows x 32 k

    const int tid  = threadIdx.x;
    const int lane = tid & 31;
    const int wave = tid >> 5;
    const int bh = blockIdx.x;
    const int b = bh >> 3, h = bh & 7;
    const int tm = wave >> 2, tn = wave & 3;

    const unsigned int* kT32 = reinterpret_cast<const unsigned int*>(kT);

    unsigned int   raA[2];
    unsigned short rbl[2], rbh[2];

    auto gload = [&](int k0) {
        #pragma unroll
        for (int j = 0; j < 2; ++j) {
            int idx = tid + (j << 9);
            int r = idx >> 4, c2 = idx & 15;
            raA[j] = kT32[((size_t)(bh * D_ + r) << 11) + (k0 >> 1) + c2];
        }
        #pragma unroll
        for (int j = 0; j < 2; ++j) {
            int idx = tid + (j << 9);
            int e = idx & 63, k2 = idx >> 6;
            rbl[j] = v_bf[(size_t)(b * N_ + k0 + (k2 << 1)    ) * C_ + h * D_ + e];
            rbh[j] = v_bf[(size_t)(b * N_ + k0 + (k2 << 1) + 1) * C_ + h * D_ + e];
        }
    };
    auto lstore = [&](int buf) {
        #pragma unroll
        for (int j = 0; j < 2; ++j) {
            int idx = tid + (j << 9);
            int r = idx >> 4, c2 = idx & 15;
            smA[buf][r * 16 + c2] = raA[j];
        }
        #pragma unroll
        for (int j = 0; j < 2; ++j) {
            int idx = tid + (j << 9);
            int e = idx & 63, k2 = idx >> 6;
            smB[buf][e * 16 + k2] =
                (unsigned int)rbl[j] | ((unsigned int)rbh[j] << 16);
        }
    };

    v8f acc = {};
    gload(0);
    lstore(0);
    __syncthreads();

    int cur = 0;
    for (int k0 = 0; k0 < N_; k0 += 32) {
        const bool more = (k0 + 32) < N_;
        if (more) gload(k0 + 32);

        v16bf a = load_frag_a(
            reinterpret_cast<const unsigned short*>(smA[cur]) + tm * 16 * 32,
            lane);
        v16bf bb = load_frag_b(
            reinterpret_cast<const unsigned short*>(smB[cur]) + tn * 16 * 32,
            lane);
        acc = __builtin_amdgcn_wmma_f32_16x16x32_bf16(
                  false, a, false, bb, (short)0, acc, false, false);

        if (more) lstore(cur ^ 1);
        __syncthreads();
        cur ^= 1;
    }
    const int col = tn * 16 + (lane & 15);
    const int hi8 = (lane >> 4) << 3;
    #pragma unroll
    for (int r = 0; r < 8; ++r) {
        int row = tm * 16 + r + hi8;
        kv_state[(size_t)bh * (D_ * D_) + row * D_ + col] = acc[r];
    }
}

// ---------------------------------------------------------------------------
// Kernel 6: out[b,n,h,e] = (q @ kv_state) * z  +  depthwise conv3(v) + bias
// One wave computes 16 rows x 64 cols (one head). K = 64 (two WMMA steps).
// ---------------------------------------------------------------------------
__global__ __launch_bounds__(256) void out_kernel(
    const unsigned short* __restrict__ q_bf,
    const float* __restrict__ kv_state,
    const float* __restrict__ z,
    const unsigned short* __restrict__ v_bf,
    const float* __restrict__ lepe_w,
    const float* __restrict__ lepe_b,
    float* __restrict__ out)
{
    __shared__ __align__(16) unsigned int smA[8][16 * 16];      // per-wave 16x32
    __shared__ __align__(16) unsigned int smB[8][4][16 * 16];   // per-wave 4 tiles

    const int lane = threadIdx.x & 31;
    const int wave = threadIdx.x >> 5;
    const int wg = blockIdx.x * 8 + wave;         // 0..16383
    const int bh = wg >> 8;
    const int tm = wg & 255;
    const int b = bh >> 3, h = bh & 7;

    const unsigned int* q32 = reinterpret_cast<const unsigned int*>(q_bf);

    v8f acc[4] = {{}, {}, {}, {}};
    for (int k0 = 0; k0 < D_; k0 += 32) {
        // A: q rows, copy as dwords (256 dwords / 32 lanes)
        #pragma unroll
        for (int j = 0; j < 8; ++j) {
            int idx = lane + (j << 5);
            int r = idx >> 4, c2 = idx & 15;
            smA[wave][r * 16 + c2] =
                q32[(size_t)(b * N_ + tm * 16 + r) * (C_ / 2) +
                    (h * D_ + k0) / 2 + c2];
        }
        // B: kv_state f32 -> bf16 pairs along k
        #pragma unroll
        for (int t = 0; t < 4; ++t) {
            #pragma unroll
            for (int j = 0; j < 8; ++j) {
                int idx = lane + (j << 5);
                int c = idx & 15, k2 = idx >> 4;
                float lo = kv_state[(size_t)bh * (D_ * D_) +
                                    (k0 + (k2 << 1)) * D_ + t * 16 + c];
                float hi = kv_state[(size_t)bh * (D_ * D_) +
                                    (k0 + (k2 << 1) + 1) * D_ + t * 16 + c];
                smB[wave][t][c * 16 + k2] = pack_bf2(lo, hi);
            }
        }
        v16bf a = load_frag_a(
            reinterpret_cast<const unsigned short*>(smA[wave]), lane);
        #pragma unroll
        for (int t = 0; t < 4; ++t) {
            v16bf bb = load_frag_b(
                reinterpret_cast<const unsigned short*>(smB[wave][t]), lane);
            acc[t] = __builtin_amdgcn_wmma_f32_16x16x32_bf16(
                         false, a, false, bb, (short)0, acc[t], false, false);
        }
    }

    const int cl  = lane & 15;
    const int hi8 = (lane >> 4) << 3;

    float zv8[8];
    #pragma unroll
    for (int r = 0; r < 8; ++r)
        zv8[r] = z[(size_t)bh * N_ + tm * 16 + r + hi8];

    #pragma unroll
    for (int t = 0; t < 4; ++t) {
        const int col = h * D_ + t * 16 + cl;     // channel 0..511
        const float w0 = lepe_w[col * 3 + 0];
        const float w1 = lepe_w[col * 3 + 1];
        const float w2 = lepe_w[col * 3 + 2];
        const float lb = lepe_b[col];
        #pragma unroll
        for (int r = 0; r < 8; ++r) {
            int n = tm * 16 + r + hi8;
            size_t vb = (size_t)b * N_;
            float o   = acc[t][r] * zv8[r];
            float vm1 = (n > 0)      ? bf2f(v_bf[(vb + n - 1) * C_ + col]) : 0.f;
            float v0  =                bf2f(v_bf[(vb + n    ) * C_ + col]);
            float vp1 = (n < N_ - 1) ? bf2f(v_bf[(vb + n + 1) * C_ + col]) : 0.f;
            // v was stored * (1/64); conv is linear -> multiply back by 64
            float lepe = 64.f * (w0 * vm1 + w1 * v0 + w2 * vp1) + lb;
            out[(vb + n) * C_ + col] = o + lepe;
        }
    }
}

// ---------------------------------------------------------------------------
// Launch
// ---------------------------------------------------------------------------
extern "C" void kernel_launch(void* const* d_in, const int* in_sizes, int n_in,
                              void* d_out, int out_size, void* d_ws, size_t ws_size,
                              hipStream_t stream)
{
    (void)in_sizes; (void)n_in; (void)out_size; (void)ws_size;
    const float* x_q    = (const float*)d_in[0];
    const float* x_kv   = (const float*)d_in[1];
    const float* Wq     = (const float*)d_in[2];
    const float* bq     = (const float*)d_in[3];
    const float* Wkv    = (const float*)d_in[4];
    const float* bkv    = (const float*)d_in[5];
    const float* lepe_w = (const float*)d_in[6];
    const float* lepe_b = (const float*)d_in[7];
    float* out = (float*)d_out;

    char* ws = (char*)d_ws;
    const size_t MB = 1024ull * 1024ull;
    unsigned short* q_bf     = (unsigned short*)(ws);                 // 32 MB
    unsigned short* kT_bf    = (unsigned short*)(ws + 32 * MB);       // 32 MB
    unsigned short* v_bf     = (unsigned short*)(ws + 64 * MB);       // 32 MB
    float*          k_mean   = (float*)(ws + 96 * MB);                // 16 KB
    float*          zbuf     = (float*)(ws + 97 * MB);                // 1 MB
    float*          kv_state = (float*)(ws + 98 * MB);                // 1 MB

    // Q projection: 256 x 4 blocks of 128x128
    proj_kernel<<<dim3(256, 4), 256, 0, stream>>>(x_q, Wq, bq, C_, 0,
                                                  q_bf, nullptr, nullptr);
    // KV projection: 256 x 8 blocks of 128x128
    proj_kernel<<<dim3(256, 8), 256, 0, stream>>>(x_kv, Wkv, bkv, 2 * C_, 1,
                                                  nullptr, kT_bf, v_bf);
    ksum_kernel<<<4096, 256, 0, stream>>>(kT_bf, k_mean);
    z_kernel<<<1024, 256, 0, stream>>>(q_bf, k_mean, zbuf);
    kvstate_kernel<<<64, 512, 0, stream>>>(kT_bf, v_bf, kv_state);
    out_kernel<<<2048, 256, 0, stream>>>(q_bf, kv_state, zbuf, v_bf,
                                         lepe_w, lepe_b, out);
}
